// MyConv2DLayer_19679540150355
// MI455X (gfx1250) — matrix operand order repaired and checked by reading, not compiled, not run on hardware
//
#include <hip/hip_runtime.h>
#include <stdint.h>

typedef __attribute__((ext_vector_type(16))) __bf16 v16bf;
typedef __attribute__((ext_vector_type(2)))  __bf16 bf16x2;
typedef __attribute__((ext_vector_type(8)))  float  v8f;
typedef __attribute__((ext_vector_type(4)))  unsigned int u32x4;
typedef __attribute__((ext_vector_type(8)))  int i32x8;
typedef __attribute__((ext_vector_type(4)))  int i32x4;

#define BATCH   32
#define OHW     62
#define FOUT    256
#define CIN     128
#define IHW     64
#define KTOT    1152   // 9 taps * 128 channels
#define BM      128
#define BN      128
#define LDA     40     // halfs per padded LDS row (80B; 64B data + 16B pad)
#define KSTEPS  36     // 1152 / 32

#if __has_builtin(__builtin_amdgcn_tensor_load_to_lds) && __has_builtin(__builtin_amdgcn_s_wait_tensorcnt)
#define HAVE_TDM 1
#else
#define HAVE_TDM 0
#endif

// ---- bf16 split helpers: native __bf16 converts (RNE) ----------------------
__device__ __forceinline__ void split_pack(float a, float b, unsigned& hi, unsigned& lo) {
  __bf16 ha = (__bf16)a, hb = (__bf16)b;
  float  ra = a - (float)ha, rb = b - (float)hb;
  bf16x2 vh = {ha, hb};
  bf16x2 vl = {(__bf16)ra, (__bf16)rb};
  hi = __builtin_bit_cast(unsigned, vh);
  lo = __builtin_bit_cast(unsigned, vl);
}

union FragU { v16bf v; uint4 q[2]; };

// Tap offset into the input for k-step ks (32 channels per step, 4 steps/tap)
__device__ __forceinline__ int a_offset(int ks) {
  const int tap = ks >> 2;
  const int kh  = tap / 3;
  const int kw  = tap - kh * 3;
  return (kh * IHW + kw) * CIN + ((ks & 3) << 5);
}

__device__ __forceinline__ void convert_store_a(unsigned short* Ah, unsigned short* Al,
                                                int arow, int ac, const float4* xa) {
  uint4 h0, l0, h1, l1;
  split_pack(xa[0].x, xa[0].y, h0.x, l0.x);
  split_pack(xa[0].z, xa[0].w, h0.y, l0.y);
  split_pack(xa[1].x, xa[1].y, h0.z, l0.z);
  split_pack(xa[1].z, xa[1].w, h0.w, l0.w);
  split_pack(xa[2].x, xa[2].y, h1.x, l1.x);
  split_pack(xa[2].z, xa[2].w, h1.y, l1.y);
  split_pack(xa[3].x, xa[3].y, h1.z, l1.z);
  split_pack(xa[3].z, xa[3].w, h1.w, l1.w);
  uint4* dh = (uint4*)&Ah[arow * LDA + ac];
  uint4* dl = (uint4*)&Al[arow * LDA + ac];
  dh[0] = h0; dh[1] = h1;
  dl[0] = l0; dl[1] = l1;
}

// ---- TDM: DMA one 128-row x 64B tile of bf16 [n][k] weights into LDS -------
// data_size=4B; tensor row = 576 dwords (1152 bf16); tile = 16 dwords x 128 rows;
// LDS padding: every 16 dwords insert 4 dwords -> 80B LDS row stride (== LDA).
#if HAVE_TDM
__device__ __forceinline__ void tdm_load_tile(unsigned lds_off, const unsigned short* gptr) {
  unsigned long long ga = (unsigned long long)(uintptr_t)gptr;
  u32x4 g0;
  g0.x = 1u;                                             // count=1 (user mode)
  g0.y = lds_off;                                        // lds_addr (bytes)
  g0.z = (unsigned)ga;                                   // global_addr[31:0]
  g0.w = (unsigned)((ga >> 32) & 0x01FFFFFFu) | 0x80000000u; // addr[56:32] | type=2
  i32x8 g1;
  g1[0] = (int)((2u << 16) | (1u << 20) | (3u << 22) | (3u << 25)); // dsize=4B, pad_en, int=16dw, amt=4dw
  g1[1] = (int)(576u << 16);        // tensor_dim0[15:0] in [31:16]
  g1[2] = (int)(256u << 16);        // tensor_dim0[31:16]=0 | tensor_dim1[15:0]
  g1[3] = (int)(16u  << 16);        // tensor_dim1[31:16]=0 | tile_dim0=16 dwords
  g1[4] = 128;                      // tile_dim1=128 rows, tile_dim2=0
  g1[5] = 576;                      // tensor_dim0_stride[31:0] = 576 dwords
  g1[6] = 0;
  g1[7] = 0;
  i32x4 z4 = {0, 0, 0, 0};
#if __clang_major__ >= 23
  i32x8 z8 = {0, 0, 0, 0, 0, 0, 0, 0};
  __builtin_amdgcn_tensor_load_to_lds(g0, g1, z4, z4, z8, 0);
#else
  __builtin_amdgcn_tensor_load_to_lds(g0, g1, z4, z4, 0);
#endif
}
#endif

// ---- prep: split weights to bf16 hi/lo, transposed to [n][k] ----------------
__global__ void wsplit_kernel(const float* __restrict__ w,
                              unsigned short* __restrict__ bhi,
                              unsigned short* __restrict__ blo) {
  int idx = blockIdx.x * blockDim.x + threadIdx.x;  // idx = k*FOUT + n
  if (idx >= KTOT * FOUT) return;
  int k = idx / FOUT;
  int n = idx - k * FOUT;
  float x = w[idx];
  __bf16 h = (__bf16)x;
  __bf16 l = (__bf16)(x - (float)h);
  bhi[n * KTOT + k] = __builtin_bit_cast(unsigned short, h);
  blo[n * KTOT + k] = __builtin_bit_cast(unsigned short, l);
}

// ---- main: pipelined implicit-GEMM conv, bf16x3 WMMA + TDM B-staging -------
__global__ __launch_bounds__(256) void conv_wmma_kernel(
    const float* __restrict__ in,
    const unsigned short* __restrict__ bhi,
    const unsigned short* __restrict__ blo,
    float* __restrict__ out)
{
  // Double-buffered LDS (ping-pong), 80 KB total
  __shared__ unsigned short As_hi[2][BM * LDA];
  __shared__ unsigned short As_lo[2][BM * LDA];
  __shared__ unsigned short Bs_hi[2][BN * LDA];
  __shared__ unsigned short Bs_lo[2][BN * LDA];

  const int t    = threadIdx.x;
  const int lane = t & 31;
  const int wid  = t >> 5;
  const int wm   = wid & 3;       // M offset = wm*32
  const int wn   = wid >> 2;      // N offset = wn*64
  const int hh   = lane >> 4;     // half-wave
  const int rr   = lane & 15;

  // A-staging coordinates (fixed per thread; only the tap offset varies)
  const int arow = t >> 1;             // 0..127 : M-row within tile
  const int ac   = (t & 1) << 4;       // 0 or 16 : channel offset
  const int m    = blockIdx.x * BM + arow;      // permuted output row index
  const int bb   = m & (BATCH - 1);
  const int pix  = m >> 5;
  const int ow_  = pix % OHW;
  const int oh_  = pix / OHW;
  const float* aptr = in + ((bb * IHW + oh_) * IHW + ow_) * CIN + ac;

  const int n0 = blockIdx.y * BN;
#if HAVE_TDM
  const unsigned lds_bhi[2] = { (unsigned)(uintptr_t)(void*)Bs_hi[0],
                                (unsigned)(uintptr_t)(void*)Bs_hi[1] };
  const unsigned lds_blo[2] = { (unsigned)(uintptr_t)(void*)Bs_lo[0],
                                (unsigned)(uintptr_t)(void*)Bs_lo[1] };
#else
  const int brow = t >> 2;             // 0..63 (two passes -> 128 rows)
  const int bq   = t & 3;              // 16B chunk within 64B row
#endif

  const v8f vzero = {0.f, 0.f, 0.f, 0.f, 0.f, 0.f, 0.f, 0.f};
  v8f acc[2][4];
#pragma unroll
  for (int i = 0; i < 2; ++i)
#pragma unroll
    for (int j = 0; j < 4; ++j) acc[i][j] = vzero;

  // ---- Prologue: stage step 0 into buffer 0 ----
  {
#if HAVE_TDM
    if (wid == 0) {
      tdm_load_tile(lds_bhi[0], bhi + n0 * KTOT);
      tdm_load_tile(lds_blo[0], blo + n0 * KTOT);
    }
#else
#pragma unroll
    for (int p = 0; p < 2; ++p) {
      const int row  = brow + (p << 6);
      const int goff = (n0 + row) * KTOT + (bq << 3);
      *(uint4*)&Bs_hi[0][row * LDA + (bq << 3)] = *(const uint4*)(bhi + goff);
      *(uint4*)&Bs_lo[0][row * LDA + (bq << 3)] = *(const uint4*)(blo + goff);
    }
#endif
    float4 xa[4];
    const float4* g = (const float4*)aptr;   // a_offset(0) == 0
    xa[0] = g[0]; xa[1] = g[1]; xa[2] = g[2]; xa[3] = g[3];
    convert_store_a(As_hi[0], As_lo[0], arow, ac, xa);
#if HAVE_TDM
    if (wid == 0) __builtin_amdgcn_s_wait_tensorcnt((short)0);
#endif
    __syncthreads();
  }

  // ---- Pipelined main loop: one barrier per k-step ----
#pragma unroll 1
  for (int ks = 0; ks < KSTEPS; ++ks) {
    const int cur  = ks & 1;
    const int nxt  = cur ^ 1;
    const bool more = (ks + 1) < KSTEPS;

    float4 xa[4];
#if !HAVE_TDM
    uint4 xbh[2], xbl[2];
#endif
    if (more) {
#if HAVE_TDM
      if (wid == 0) {
        const int goff = n0 * KTOT + ((ks + 1) << 5);
        tdm_load_tile(lds_bhi[nxt], bhi + goff);
        tdm_load_tile(lds_blo[nxt], blo + goff);
      }
#else
      const int kbase = (ks + 1) << 5;
#pragma unroll
      for (int p = 0; p < 2; ++p) {
        const int goff = (n0 + brow + (p << 6)) * KTOT + kbase + (bq << 3);
        xbh[p] = *(const uint4*)(bhi + goff);
        xbl[p] = *(const uint4*)(blo + goff);
      }
#endif
      const float4* g = (const float4*)(aptr + a_offset(ks + 1));
      xa[0] = g[0]; xa[1] = g[1]; xa[2] = g[2]; xa[3] = g[3];
      if (ks + 2 < KSTEPS) __builtin_prefetch(aptr + a_offset(ks + 2), 0, 1);
    }

    // ---- Compute step ks out of buffer `cur` ----
    const unsigned short* Ah = As_hi[cur];
    const unsigned short* Al = As_lo[cur];
    const unsigned short* Bh = Bs_hi[cur];
    const unsigned short* Bl = Bs_lo[cur];

    FragU a_hi[2], a_lo[2], b_hi[4], b_lo[4];
#pragma unroll
    for (int mt = 0; mt < 2; ++mt) {
      const int off = (wm * 32 + mt * 16 + rr) * LDA + (hh << 3);
      a_hi[mt].q[0] = *(const uint4*)&Ah[off];
      a_hi[mt].q[1] = *(const uint4*)&Ah[off + 16];
      a_lo[mt].q[0] = *(const uint4*)&Al[off];
      a_lo[mt].q[1] = *(const uint4*)&Al[off + 16];
    }
#pragma unroll
    for (int nt = 0; nt < 4; ++nt) {
      const int off = (wn * 64 + nt * 16 + rr) * LDA + (hh << 3);
      b_hi[nt].q[0] = *(const uint4*)&Bh[off];
      b_hi[nt].q[1] = *(const uint4*)&Bh[off + 16];
      b_lo[nt].q[0] = *(const uint4*)&Bl[off];
      b_lo[nt].q[1] = *(const uint4*)&Bl[off + 16];
    }

    // bf16x3: hi*hi + hi*lo + lo*hi (lo*lo dropped, ~2^-16 relative)
#pragma unroll
    for (int mt = 0; mt < 2; ++mt)
#pragma unroll
      for (int nt = 0; nt < 4; ++nt) {
        v8f c = acc[mt][nt];
        c = __builtin_amdgcn_wmma_f32_16x16x32_bf16(false, a_hi[mt].v, false, b_hi[nt].v, (short)0, c, false, false);
        c = __builtin_amdgcn_wmma_f32_16x16x32_bf16(false, a_hi[mt].v, false, b_lo[nt].v, (short)0, c, false, false);
        c = __builtin_amdgcn_wmma_f32_16x16x32_bf16(false, a_lo[mt].v, false, b_hi[nt].v, (short)0, c, false, false);
        acc[mt][nt] = c;
      }

    // ---- Drain staging for step ks+1 into buffer `nxt` ----
    if (more) {
      convert_store_a(As_hi[nxt], As_lo[nxt], arow, ac, xa);
#if !HAVE_TDM
#pragma unroll
      for (int p = 0; p < 2; ++p) {
        const int row = brow + (p << 6);
        *(uint4*)&Bs_hi[nxt][row * LDA + (bq << 3)] = xbh[p];
        *(uint4*)&Bs_lo[nxt][row * LDA + (bq << 3)] = xbl[p];
      }
#endif
#if HAVE_TDM
      if (wid == 0) __builtin_amdgcn_s_wait_tensorcnt((short)0);
#endif
    }
    __syncthreads();
  }

  // Epilogue: ReLU + store. acc VGPR i holds (M = i + 8h, N = rr) of each 16x16 tile.
  const int mbase = blockIdx.x * BM + wm * 32 + (hh << 3);
  const int fbase = blockIdx.y * BN + wn * 64 + rr;
#pragma unroll
  for (int mt = 0; mt < 2; ++mt)
#pragma unroll
    for (int nt = 0; nt < 4; ++nt) {
      const int mrow = mbase + mt * 16;
      const int fcol = fbase + nt * 16;
#pragma unroll
      for (int i = 0; i < 8; ++i)
        out[(size_t)(mrow + i) * FOUT + fcol] = fmaxf(acc[mt][nt][i], 0.f);
    }
}

extern "C" void kernel_launch(void* const* d_in, const int* in_sizes, int n_in,
                              void* d_out, int out_size, void* d_ws, size_t ws_size,
                              hipStream_t stream) {
  (void)in_sizes; (void)n_in; (void)out_size; (void)ws_size;
  const float* in = (const float*)d_in[0];
  const float* wk = (const float*)d_in[1];
  float* out = (float*)d_out;

  unsigned short* bhi = (unsigned short*)d_ws;                 // 1152*256 bf16
  unsigned short* blo = bhi + (size_t)KTOT * FOUT;             // +1152*256 bf16 (~1.2 MB total)

  wsplit_kernel<<<(KTOT * FOUT + 255) / 256, 256, 0, stream>>>(wk, bhi, blo);

  dim3 grid(961, 2);   // M = 123008 = 961*128 exactly; N = 256 = 2*128
  conv_wmma_kernel<<<grid, 256, 0, stream>>>(in, bhi, blo, out);
}
